// GCPNPolicy_55155970016019
// MI455X (gfx1250) — matrix operand order repaired and checked by reading, not compiled, not run on hardware
//
#include <hip/hip_runtime.h>
#include <cstdint>
#include <cstddef>

#define HDIM 256
#define DIN  128

typedef __attribute__((ext_vector_type(16))) __bf16 v16bf;
typedef __attribute__((ext_vector_type(8)))  float  v8f;
typedef __attribute__((ext_vector_type(8)))  unsigned short ushort8;
typedef __attribute__((ext_vector_type(4)))  unsigned int   uint4v;
typedef int i32x4g __attribute__((vector_size(16)));   // matches async builtin param pointee

__device__ __forceinline__ unsigned short f2bf(float x) {
    unsigned int u = __float_as_uint(x);
    u += 0x7FFFu + ((u >> 16) & 1u);          // round-to-nearest-even
    return (unsigned short)(u >> 16);
}

// ---------------------------------------------------------------------------
// WMMA bf16 GEMM:  C[M x NCOLS] = A[M x K] * Bt^T  (+ bias, optional relu)
// A: bf16 row-major [M x K];  Bt: bf16 row-major [NCOLS x K]  (i.e. W^T)
// Block = 256 threads (8 waves), computes 16 rows x NCOLS.
// NCOLS is a template param so the per-wave tile count (and acc[] indexing)
// is fully compile-time -> no indirect VGPR addressing around the WMMAs.
// Tiles staged in LDS via async global->LDS b128 copies when available.
// ---------------------------------------------------------------------------
#if defined(__gfx1250__) && __has_builtin(__builtin_amdgcn_global_load_async_to_lds_b128)
#define ASYNC_COPY 1
#else
#define ASYNC_COPY 0
#endif

template <int NCOLS>
__global__ __launch_bounds__(256) void gemm_bf16_wmma(
    const unsigned short* __restrict__ A, const unsigned short* __restrict__ Bt,
    const float* __restrict__ bias, float* __restrict__ C,
    int M, int K, int relu)
{
    constexpr int TPW = NCOLS / 128;               // n-tiles per wave: 2 (256) or 1 (128)
    __shared__ __align__(16) unsigned short ldsA[16 * 32];
    __shared__ __align__(16) unsigned short ldsB[NCOLS * 32];
    const int tid  = threadIdx.x;
    const int lane = tid & 31;
    const int wave = tid >> 5;
    const int m0   = blockIdx.x * 16;

    v8f acc[TPW];
#pragma unroll
    for (int i = 0; i < TPW; ++i)
#pragma unroll
        for (int j = 0; j < 8; ++j) acc[i][j] = 0.f;

    constexpr int total_xf = 64 + NCOLS * 4;       // 16B transfers: A = 16 rows*4, B = NCOLS*4

    for (int kc = 0; kc < K; kc += 32) {
#pragma unroll 2
        for (int c = tid; c < total_xf; c += 256) {
            const unsigned short* src;
            unsigned short* dst;
            if (c < 64) {
                int r = c >> 2, q = c & 3;
                src = A + (size_t)(m0 + r) * K + kc + q * 8;
                dst = ldsA + r * 32 + q * 8;
            } else {
                int cb = c - 64;
                int n = cb >> 2, q = cb & 3;
                src = Bt + (size_t)n * K + kc + q * 8;
                dst = ldsB + n * 32 + q * 8;
            }
#if ASYNC_COPY
            unsigned short* s_nc = const_cast<unsigned short*>(src);
            __builtin_amdgcn_global_load_async_to_lds_b128(
                (__attribute__((address_space(1))) i32x4g*)s_nc,
                (__attribute__((address_space(3))) i32x4g*)dst, 0, 0);
#else
            *(uint4v*)dst = *(const uint4v*)src;
#endif
        }
#if ASYNC_COPY
#if __has_builtin(__builtin_amdgcn_s_wait_asynccnt)
        __builtin_amdgcn_s_wait_asynccnt(0);
#else
        asm volatile("s_wait_asynccnt 0" ::: "memory");
#endif
#endif
        __syncthreads();

        // A fragment, 16x32 bf16 (ISA 7.12.2): row m = lane&15,
        // elements 0..7 -> K = ko..ko+7, elements 8..15 -> K = ko+16..ko+23,
        // ko = (lane>>4)*8
        const int am = lane & 15;
        const int ko = (lane >> 4) * 8;
        union { v16bf v; ushort8 h[2]; } af;
        af.h[0] = *(const ushort8*)&ldsA[am * 32 + ko];
        af.h[1] = *(const ushort8*)&ldsA[am * 32 + ko + 16];

#pragma unroll
        for (int ti = 0; ti < TPW; ++ti) {
            // B fragment, 32x16 bf16: elements 0..7 K=kb..kb+7, 8..15 K=kb+16..kb+23, kb=(lane>>4)*8,
            // column n = lane&15 (Bt tile stored [n][k], contiguous in k)
            const int t  = wave + ti * 8;
            const int bn = t * 16 + (lane & 15);
            const int kb = (lane >> 4) * 8;
            union { v16bf v; ushort8 h[2]; } bf_;
            bf_.h[0] = *(const ushort8*)&ldsB[bn * 32 + kb];
            bf_.h[1] = *(const ushort8*)&ldsB[bn * 32 + kb + 16];
            acc[ti] = __builtin_amdgcn_wmma_f32_16x16x32_bf16(
                false, af.v, false, bf_.v, (short)0, acc[ti], false, false);
        }
        __syncthreads();
    }

    // C/D layout: element v, lane l -> row m = v + 8*(l>>4), col n = l&15
#pragma unroll
    for (int ti = 0; ti < TPW; ++ti) {
        const int t  = wave + ti * 8;
        const int n  = t * 16 + (lane & 15);
        const int mr = (lane >> 4) * 8;
        const float bv = bias ? bias[n] : 0.f;
#pragma unroll
        for (int v = 0; v < 8; ++v) {
            float val = acc[ti][v] + bv;
            if (relu) val = fmaxf(val, 0.f);
            C[(size_t)(m0 + mr + v) * NCOLS + n] = val;
        }
    }
}

// ---------------------------------------------------------------------------
// Elementwise / helper kernels
// ---------------------------------------------------------------------------
__global__ void fill_zero(float* p, long long n) {
    long long i = (long long)blockIdx.x * blockDim.x + threadIdx.x;
    if (i < n) p[i] = 0.f;
}

__global__ void f32_to_bf16_k(const float* __restrict__ s, unsigned short* __restrict__ d, long long n) {
    long long i = (long long)blockIdx.x * blockDim.x + threadIdx.x;
    if (i < n) d[i] = f2bf(s[i]);
}

// W [K x Nc] f32 row-major  ->  Wt [Nc x K] bf16 row-major
__global__ void transpose_to_bf16(const float* __restrict__ w, unsigned short* __restrict__ wt, int K, int Nc) {
    int idx = blockIdx.x * blockDim.x + threadIdx.x;
    if (idx >= K * Nc) return;
    int n = idx / K, k = idx - n * K;
    wt[idx] = f2bf(w[(size_t)k * Nc + n]);
}

__global__ void deg_init(float* deg, int N) {
    int i = blockIdx.x * blockDim.x + threadIdx.x;
    if (i < N) deg[i] = 1.f;                     // self loop
}
__global__ void deg_edges(const int* __restrict__ dst, float* deg, int E) {
    int e = blockIdx.x * blockDim.x + threadIdx.x;
    if (e < E)
        __hip_atomic_fetch_add(&deg[dst[e]], 1.f, __ATOMIC_RELAXED, __HIP_MEMORY_SCOPE_AGENT);
}
__global__ void deg_fin(float* deg, int N) {
    int i = blockIdx.x * blockDim.x + threadIdx.x;
    if (i < N) deg[i] = rsqrtf(fmaxf(deg[i], 1.f));
}

// agg[i][c] = bias[c] + dinv[i]^2 * tmp[i][c]        (self-loop term)
__global__ void gcn_self_bias(const float* __restrict__ tmp, const float* __restrict__ dinv,
                              const float* __restrict__ bias, float* __restrict__ agg, int N) {
    long long idx = (long long)blockIdx.x * blockDim.x + threadIdx.x;
    if (idx >= (long long)N * HDIM) return;
    int i = (int)(idx >> 8), c = (int)(idx & 255);
    float di = dinv[i];
    agg[idx] = bias[c] + di * di * tmp[idx];
}

// one wave per edge: agg[d] += dinv[s]*dinv[d] * tmp[s]
__global__ __launch_bounds__(256) void gcn_edges(
    const float* __restrict__ tmp, const int* __restrict__ src, const int* __restrict__ dst,
    const float* __restrict__ dinv, float* __restrict__ agg, int E)
{
    int wid  = (int)(((long long)blockIdx.x * blockDim.x + threadIdx.x) >> 5);
    int lane = threadIdx.x & 31;
    if (wid >= E) return;
    int s = src[wid], d = dst[wid];
    float coef = dinv[s] * dinv[d];
    const float* hs = tmp + (size_t)s * HDIM;
    float* od = agg + (size_t)d * HDIM;
#pragma unroll
    for (int j = 0; j < 8; ++j) {
        int c = lane + 32 * j;
        __hip_atomic_fetch_add(&od[c], coef * hs[c], __ATOMIC_RELAXED, __HIP_MEMORY_SCOPE_AGENT);
    }
}

// out[row] = relu(LN(agg[row])*g + b) + res[row]   — one wave32 per row
__global__ __launch_bounds__(256) void ln_relu_res(
    const float* __restrict__ agg, const float* __restrict__ g, const float* __restrict__ b,
    const float* __restrict__ res, float* __restrict__ out, int N)
{
    int row  = blockIdx.x * 8 + (threadIdx.x >> 5);
    int lane = threadIdx.x & 31;
    if (row >= N) return;
    const float* h = agg + (size_t)row * HDIM;
    float v[8], s = 0.f;
#pragma unroll
    for (int j = 0; j < 8; ++j) { v[j] = h[lane + 32 * j]; s += v[j]; }
    for (int o = 16; o > 0; o >>= 1) s += __shfl_xor(s, o, 32);
    float mean = s * (1.f / HDIM);
    float var = 0.f;
#pragma unroll
    for (int j = 0; j < 8; ++j) { float d = v[j] - mean; var += d * d; }
    for (int o = 16; o > 0; o >>= 1) var += __shfl_xor(var, o, 32);
    float rs = rsqrtf(var * (1.f / HDIM) + 1e-5f);
#pragma unroll
    for (int j = 0; j < 8; ++j) {
        int c = lane + 32 * j;
        float val = (v[j] - mean) * rs * g[c] + b[c];
        val = fmaxf(val, 0.f);
        out[(size_t)row * HDIM + c] = val + res[(size_t)row * HDIM + c];
    }
}

__global__ void pool_count(const int* __restrict__ batch, float* cnt, int N) {
    int i = blockIdx.x * blockDim.x + threadIdx.x;
    if (i < N)
        __hip_atomic_fetch_add(&cnt[batch[i]], 1.f, __ATOMIC_RELAXED, __HIP_MEMORY_SCOPE_AGENT);
}
__global__ __launch_bounds__(256) void pool_sum(
    const float* __restrict__ emb, const int* __restrict__ batch, float* pool, int N)
{
    int wid  = (int)(((long long)blockIdx.x * blockDim.x + threadIdx.x) >> 5);
    int lane = threadIdx.x & 31;
    if (wid >= N) return;
    int g = batch[wid];
    const float* h = emb + (size_t)wid * HDIM;
    float* p = pool + (size_t)g * HDIM;
#pragma unroll
    for (int j = 0; j < 8; ++j) {
        int c = lane + 32 * j;
        __hip_atomic_fetch_add(&p[c], h[c], __ATOMIC_RELAXED, __HIP_MEMORY_SCOPE_AGENT);
    }
}
__global__ void pool_div(const float* pool, const float* cnt, float* gemb, int B) {
    int idx = blockIdx.x * blockDim.x + threadIdx.x;
    if (idx >= B * HDIM) return;
    gemb[idx] = pool[idx] / fmaxf(cnt[idx >> 8], 1.f);
}

__global__ void gather_rows_bf16(const float* __restrict__ emb, const int* __restrict__ rows,
                                 unsigned short* __restrict__ dst, int R) {
    int idx = blockIdx.x * blockDim.x + threadIdx.x;
    if (idx >= R * HDIM) return;
    int r = idx >> 8, c = idx & 255;
    dst[idx] = f2bf(emb[(size_t)rows[r] * HDIM + c]);
}

// pair[i] = concat(emb[i], emb[nn[batch[i]]])   as bf16 [N x 512]
__global__ void build_pair_bf16(const float* __restrict__ emb, const int* __restrict__ batch,
                                const int* __restrict__ nn, unsigned short* __restrict__ dst, int N) {
    long long idx = (long long)blockIdx.x * blockDim.x + threadIdx.x;
    if (idx >= (long long)N * 512) return;
    int i = (int)(idx >> 9), c = (int)(idx & 511);
    float v = (c < HDIM) ? emb[(size_t)i * HDIM + c]
                         : emb[(size_t)nn[batch[i]] * HDIM + (c - HDIM)];
    dst[idx] = f2bf(v);
}

// small dense (f32 weights): out[m][n] = relu?(A[m]·W[:,n] + b[n])
__global__ void dense_small(const float* __restrict__ A, const float* __restrict__ W,
                            const float* __restrict__ b, float* __restrict__ out,
                            int M, int K, int Nc, int relu) {
    long long idx = (long long)blockIdx.x * blockDim.x + threadIdx.x;
    if (idx >= (long long)M * Nc) return;
    int m = (int)(idx / Nc), n = (int)(idx - (long long)m * Nc);
    float acc = b ? b[n] : 0.f;
    const float* a = A + (size_t)m * K;
    for (int k = 0; k < K; ++k) acc = fmaf(a[k], W[(size_t)k * Nc + n], acc);
    if (relu) acc = fmaxf(acc, 0.f);
    out[idx] = acc;
}

__global__ void copy_f32(const float* s, float* d, long long n) {
    long long i = (long long)blockIdx.x * blockDim.x + threadIdx.x;
    if (i < n) d[i] = s[i];
}

// ---------------------------------------------------------------------------
// Host launcher
// ---------------------------------------------------------------------------
static inline int cdiv(long long a, int b) { return (int)((a + b - 1) / b); }

extern "C" void kernel_launch(void* const* d_in, const int* in_sizes, int n_in,
                              void* d_out, int out_size, void* d_ws, size_t ws_size,
                              hipStream_t stream) {
    const float* x      = (const float*)d_in[0];
    const int*   eidx   = (const int*)d_in[1];
    const int*   batch  = (const int*)d_in[2];
    const int*   nn_idx = (const int*)d_in[3];
    const int*   fc_idx = (const int*)d_in[4];
    const float* conv_w[3] = { (const float*)d_in[6],  (const float*)d_in[8],  (const float*)d_in[10] };
    const float* conv_b[3] = { (const float*)d_in[7],  (const float*)d_in[9],  (const float*)d_in[11] };
    const float* ln_g[3] = { (const float*)d_in[12], (const float*)d_in[14], (const float*)d_in[16] };
    const float* ln_b[3] = { (const float*)d_in[13], (const float*)d_in[15], (const float*)d_in[17] };
    const float* res_w0 = (const float*)d_in[18]; const float* res_b0 = (const float*)d_in[19];
    const float* stop_w1 = (const float*)d_in[20]; const float* stop_b1 = (const float*)d_in[21];
    const float* stop_w2 = (const float*)d_in[22]; const float* stop_b2 = (const float*)d_in[23];
    const float* addn_w1 = (const float*)d_in[24]; const float* addn_b1 = (const float*)d_in[25];
    const float* addn_w2 = (const float*)d_in[26]; const float* addn_b2 = (const float*)d_in[27];
    const float* addb_w1 = (const float*)d_in[28]; const float* addb_b1 = (const float*)d_in[29];
    const float* addb_w2 = (const float*)d_in[30]; const float* addb_b2 = (const float*)d_in[31];
    const float* edge_w1 = (const float*)d_in[32]; const float* edge_b1 = (const float*)d_in[33];
    const float* edge_w2 = (const float*)d_in[34]; const float* edge_b2 = (const float*)d_in[35];
    const float* edge_w3 = (const float*)d_in[36]; const float* edge_b3 = (const float*)d_in[37];
    const float* bond_w1 = (const float*)d_in[38]; const float* bond_b1 = (const float*)d_in[39];
    const float* bond_w2 = (const float*)d_in[40]; const float* bond_b2 = (const float*)d_in[41];
    const float* bond_w3 = (const float*)d_in[42]; const float* bond_b3 = (const float*)d_in[43];

    const int N = in_sizes[0] / DIN;
    const int E = in_sizes[1] / 2;
    const int B = in_sizes[3];
    const int* src = eidx;
    const int* dst = eidx + E;

    // ---- workspace carve-up (256B aligned) ----
    char* ws = (char*)d_ws;
    size_t off = 0;
    auto alloc = [&](size_t bytes) { char* p = ws + off; off = (off + bytes + 255) & ~(size_t)255; return p; };
    float* cur  = (float*)alloc((size_t)N * HDIM * 4);
    float* tmp  = (float*)alloc((size_t)N * HDIM * 4);
    float* agg  = (float*)alloc((size_t)N * HDIM * 4);
    float* res0 = (float*)alloc((size_t)N * HDIM * 4);     // reused later: hid2 | h1bf
    unsigned short* abf = (unsigned short*)alloc((size_t)N * 512 * 2);
    unsigned short* wbf = (unsigned short*)alloc((size_t)720896 * 2);
    float* deg  = (float*)alloc((size_t)N * 4);
    float* pool = (float*)alloc((size_t)B * HDIM * 4);
    float* cnt  = (float*)alloc((size_t)B * 4);
    float* gemb = (float*)alloc((size_t)B * HDIM * 4);
    unsigned short* gembbf  = (unsigned short*)alloc((size_t)B * HDIM * 2);
    unsigned short* focusbf = (unsigned short*)alloc((size_t)B * HDIM * 2);
    float* headhid = (float*)alloc((size_t)B * HDIM * 4);
    // aliases into res0 (free after layer 0)
    float* hid2 = res0;                                       // [N x 128] f32
    unsigned short* h1bf = (unsigned short*)((char*)res0 + (size_t)N * 512); // [N x 256] bf16

    // pre-transposed bf16 weights (element offsets into wbf)
    unsigned short* cw_t[3] = { wbf + 0, wbf + 32768, wbf + 98304 };
    unsigned short* resw_t  = wbf + 163840;
    unsigned short* stopw_t = wbf + 196608;
    unsigned short* addnw_t = wbf + 262144;
    unsigned short* addbw_t = wbf + 327680;
    unsigned short* edgew1t = wbf + 393216;
    unsigned short* edgew2t = wbf + 524288;
    unsigned short* bondw1t = wbf + 557056;
    unsigned short* bondw2t = wbf + 688128;

    float* out      = (float*)d_out;
    float* out_stop = out;
    float* out_addn = out + (size_t)B;
    float* out_addb = out_addn + (size_t)B * 40;
    float* out_emb  = out_addb + (size_t)B * 5;
    float* out_edge = out_emb + (size_t)N * HDIM;
    float* out_bond = out_edge + (size_t)N;

    const int T = 256;
    // ---- 1. convert+transpose weights to bf16 ----
    transpose_to_bf16<<<cdiv(128 * 256, T), T, 0, stream>>>(conv_w[0], cw_t[0], 128, 256);
    transpose_to_bf16<<<cdiv(256 * 256, T), T, 0, stream>>>(conv_w[1], cw_t[1], 256, 256);
    transpose_to_bf16<<<cdiv(256 * 256, T), T, 0, stream>>>(conv_w[2], cw_t[2], 256, 256);
    transpose_to_bf16<<<cdiv(128 * 256, T), T, 0, stream>>>(res_w0, resw_t, 128, 256);
    transpose_to_bf16<<<cdiv(256 * 256, T), T, 0, stream>>>(stop_w1, stopw_t, 256, 256);
    transpose_to_bf16<<<cdiv(256 * 256, T), T, 0, stream>>>(addn_w1, addnw_t, 256, 256);
    transpose_to_bf16<<<cdiv(256 * 256, T), T, 0, stream>>>(addb_w1, addbw_t, 256, 256);
    transpose_to_bf16<<<cdiv(512 * 256, T), T, 0, stream>>>(edge_w1, edgew1t, 512, 256);
    transpose_to_bf16<<<cdiv(256 * 128, T), T, 0, stream>>>(edge_w2, edgew2t, 256, 128);
    transpose_to_bf16<<<cdiv(512 * 256, T), T, 0, stream>>>(bond_w1, bondw1t, 512, 256);
    transpose_to_bf16<<<cdiv(256 * 128, T), T, 0, stream>>>(bond_w2, bondw2t, 256, 128);

    // ---- 2. symmetric degree normalization ----
    deg_init<<<cdiv(N, T), T, 0, stream>>>(deg, N);
    deg_edges<<<cdiv(E, T), T, 0, stream>>>(dst, deg, E);
    deg_fin<<<cdiv(N, T), T, 0, stream>>>(deg, N);

    // ---- 3. GCN layers ----
    for (int l = 0; l < 3; ++l) {
        const float* hin = (l == 0) ? x : cur;
        const int K = (l == 0) ? DIN : HDIM;
        f32_to_bf16_k<<<cdiv((long long)N * K, T), T, 0, stream>>>(hin, abf, (long long)N * K);
        gemm_bf16_wmma<256><<<N / 16, T, 0, stream>>>(abf, cw_t[l], nullptr, tmp, N, K, 0);
        if (l == 0)  // residual projection shares the same bf16 A
            gemm_bf16_wmma<256><<<N / 16, T, 0, stream>>>(abf, resw_t, res_b0, res0, N, DIN, 0);
        gcn_self_bias<<<cdiv((long long)N * HDIM, T), T, 0, stream>>>(tmp, deg, conv_b[l], agg, N);
        gcn_edges<<<cdiv((long long)E * 32, T), T, 0, stream>>>(tmp, src, dst, deg, agg, E);
        ln_relu_res<<<cdiv(N, 8), T, 0, stream>>>(agg, ln_g[l], ln_b[l], (l == 0) ? res0 : cur, cur, N);
    }
    copy_f32<<<cdiv((long long)N * HDIM, T), T, 0, stream>>>(cur, out_emb, (long long)N * HDIM);

    // ---- 4. global mean pool ----
    fill_zero<<<cdiv((long long)B * HDIM, T), T, 0, stream>>>(pool, (long long)B * HDIM);
    fill_zero<<<cdiv(B, T), T, 0, stream>>>(cnt, B);
    pool_count<<<cdiv(N, T), T, 0, stream>>>(batch, cnt, N);
    pool_sum<<<cdiv((long long)N * 32, T), T, 0, stream>>>(cur, batch, pool, N);
    pool_div<<<cdiv(B * HDIM, T), T, 0, stream>>>(pool, cnt, gemb, B);

    // ---- 5. stop / add-node heads on graph embedding ----
    f32_to_bf16_k<<<cdiv((long long)B * HDIM, T), T, 0, stream>>>(gemb, gembbf, (long long)B * HDIM);
    gemm_bf16_wmma<256><<<B / 16, T, 0, stream>>>(gembbf, stopw_t, stop_b1, headhid, B, HDIM, 1);
    dense_small<<<cdiv(B, T), T, 0, stream>>>(headhid, stop_w2, stop_b2, out_stop, B, HDIM, 1, 0);
    gemm_bf16_wmma<256><<<B / 16, T, 0, stream>>>(gembbf, addnw_t, addn_b1, headhid, B, HDIM, 1);
    dense_small<<<cdiv((long long)B * 40, T), T, 0, stream>>>(headhid, addn_w2, addn_b2, out_addn, B, HDIM, 40, 0);

    // ---- 6. add-bond head on focus nodes ----
    gather_rows_bf16<<<cdiv(B * HDIM, T), T, 0, stream>>>(cur, fc_idx, focusbf, B);
    gemm_bf16_wmma<256><<<B / 16, T, 0, stream>>>(focusbf, addbw_t, addb_b1, headhid, B, HDIM, 1);
    dense_small<<<cdiv((long long)B * 5, T), T, 0, stream>>>(headhid, addb_w2, addb_b2, out_addb, B, HDIM, 5, 0);

    // ---- 7. pair MLPs (edge selection + bond type) ----
    build_pair_bf16<<<cdiv((long long)N * 512, T), T, 0, stream>>>(cur, batch, nn_idx, abf, N);
    // edge branch
    gemm_bf16_wmma<256><<<N / 16, T, 0, stream>>>(abf, edgew1t, edge_b1, tmp, N, 512, 1);
    f32_to_bf16_k<<<cdiv((long long)N * HDIM, T), T, 0, stream>>>(tmp, h1bf, (long long)N * HDIM);
    gemm_bf16_wmma<128><<<N / 16, T, 0, stream>>>(h1bf, edgew2t, edge_b2, hid2, N, HDIM, 1);
    dense_small<<<cdiv(N, T), T, 0, stream>>>(hid2, edge_w3, edge_b3, out_edge, N, 128, 1, 0);
    // bond branch
    gemm_bf16_wmma<256><<<N / 16, T, 0, stream>>>(abf, bondw1t, bond_b1, tmp, N, 512, 1);
    f32_to_bf16_k<<<cdiv((long long)N * HDIM, T), T, 0, stream>>>(tmp, h1bf, (long long)N * HDIM);
    gemm_bf16_wmma<128><<<N / 16, T, 0, stream>>>(h1bf, bondw2t, bond_b2, hid2, N, HDIM, 1);
    dense_small<<<cdiv((long long)N * 5, T), T, 0, stream>>>(hid2, bond_w3, bond_b3, out_bond, N, 128, 5, 0);

    (void)n_in; (void)out_size; (void)ws_size;
}
